// SelfAttention_88811333747054
// MI455X (gfx1250) — matrix operand order repaired and checked by reading, not compiled, not run on hardware
//
#include <hip/hip_runtime.h>

// Performer / FAVOR+ forward for MI455X (gfx1250, wave32, WMMA).
// Pipeline (all GEMMs via v_wmma_f32_16x16x32_bf16, f32 accumulation):
//   1. cast x, Wq/Wk/Wv/Wo, proj -> bf16 (proj zero-padded 266->288 rows)
//   2. qs = bf16(64^-0.25 * x@Wq), ks = likewise, v = bf16(x@Wv)
//   3. diag_q/k[row,h] = 0.5 * sum_d qs^2   (normalizer already folded in)
//   4. q features: per-(b,h) GEMM qs x proj^T, row-max (2-pass, in-register),
//      qp = ratio*(exp(dd - diag - rowmax)+eps) stored bf16 [b,h,n,288]
//   5. k features: global max via ordered-uint atomicMax, then exp pass
//   6. ksum[b,h,m] = sum_n kp ; context[b,h,m,d] = kp^T @ v (WMMA, K=4096)
//   7. dinv = 1/(qp . ksum) ; attn = (qp @ context) * dinv -> bf16 [row,512]
//   8. out = attn @ Wo + bo (f32)
// Feature-map traffic (2x144MB bf16 qp/kp + rereads) dominates: ~40us @23.3TB/s.
// data_dash is recomputed instead of stored in f32 (trades ~18 GF for ~560MB).

#define B_    8
#define N_    4096
#define DIM_  512
#define H_    8
#define DH_   64
#define M_    266
#define MP_   288            // padded feature dim: multiple of 32 for WMMA K
#define ROWS_ (B_ * N_)      // 32768

typedef __attribute__((ext_vector_type(16))) __bf16 v16bf;
typedef __attribute__((ext_vector_type(8)))  float  v8f;

// ---------------- WMMA helpers (CDNA5 16x16x32 bf16, wave32) ----------------

__device__ __forceinline__ v8f wmma_bf16(v16bf a, v16bf b, v8f c) {
  // (neg_a, A, neg_b, B, c_mod, C, reuse_a, reuse_b)
  return __builtin_amdgcn_wmma_f32_16x16x32_bf16(false, a, false, b, (short)0, c,
                                                 false, false);
}

// A fragment: 16x32 (MxK). Lane<16: M=lane, K-pairs {0,1},{2,3},{4,5},{6,7},
// then {16..23}. Lane>=16: M=lane-16, K offset +8 (ISA 7.12.2).
// Element (m,k) lives at p[m*mStride + k*kStride].
__device__ __forceinline__ v16bf load_a(const __bf16* p, int mStride, int kStride) {
  int lane = threadIdx.x & 31;
  int m = lane & 15;
  int koff = (lane & 16) ? 8 : 0;
  const __bf16* q = p + (size_t)m * mStride;
  v16bf f;
#pragma unroll
  for (int i = 0; i < 8; ++i) {
    int k = (i < 4) ? (koff + 2 * i) : (8 + koff + 2 * i);  // i>=4 -> 16+koff+2(i-4)
    f[2 * i]     = q[(size_t)k * kStride];
    f[2 * i + 1] = q[(size_t)(k + 1) * kStride];
  }
  return f;
}

// B fragment: 32x16 (KxN). Lane<16: N=lane, K=0..7,16..23; lane>=16: N=lane-16, K=8..15,24..31.
// Element (k,n) lives at p[k*kStride + n*nStride].
__device__ __forceinline__ v16bf load_b(const __bf16* p, int kStride, int nStride) {
  int lane = threadIdx.x & 31;
  int n = lane & 15;
  int kblk = (lane & 16) ? 8 : 0;
  const __bf16* q = p + (size_t)n * nStride + (size_t)kblk * kStride;
  v16bf f;
#pragma unroll
  for (int j = 0; j < 16; ++j) f[j] = q[(size_t)(j < 8 ? j : j + 8) * kStride];
  return f;
}

// C/D 16x16 f32: VGPR r -> row r (lanes 0-15) / row r+8 (lanes 16-31), col = lane&15.
__device__ __forceinline__ int c_row(int r) {
  return ((threadIdx.x & 31) < 16) ? r : (r + 8);
}
__device__ __forceinline__ int c_col() { return threadIdx.x & 15; }

// Order-preserving float<->uint for atomicMax over signed floats.
__device__ __forceinline__ unsigned f2ord(float f) {
  unsigned u = __float_as_uint(f);
  return (u & 0x80000000u) ? ~u : (u | 0x80000000u);
}
__device__ __forceinline__ float ord2f(unsigned o) {
  unsigned u = (o & 0x80000000u) ? (o & 0x7fffffffu) : ~o;
  return __uint_as_float(u);
}

// ---------------- conversion / init kernels ----------------

__global__ void cast_bf16_kernel(const float* __restrict__ in,
                                 __bf16* __restrict__ out, size_t n) {
  size_t i = (size_t)blockIdx.x * blockDim.x + threadIdx.x;
  if (i < n) out[i] = (__bf16)in[i];
}

__global__ void proj_pack_kernel(const float* __restrict__ proj,
                                 __bf16* __restrict__ out) {
  int i = blockIdx.x * blockDim.x + threadIdx.x;  // MP_*DH_
  if (i >= MP_ * DH_) return;
  int m = i / DH_;
  out[i] = (m < M_) ? (__bf16)proj[i] : (__bf16)0.0f;
}

__global__ void init_kmax_kernel(unsigned* kmax_u) {
  if (threadIdx.x == 0 && blockIdx.x == 0) *kmax_u = 0u;  // < ord of any real
}

// ---------------- 1) projection GEMMs: [32768,512] @ [512,512] ----------------
// Block = 256 thr (8 waves); wave -> 16 rows x 64 cols; grid (256, 8).

template <bool STORE_BF16>
__global__ void proj_gemm_kernel(const __bf16* __restrict__ A,
                                 const __bf16* __restrict__ W,
                                 void* __restrict__ OUT,
                                 const float* __restrict__ bias, float alpha) {
  int waveId  = threadIdx.x >> 5;
  int rowBase = blockIdx.x * 128 + waveId * 16;
  int colBase = blockIdx.y * 64;
  v8f c[4] = {};
  for (int kb = 0; kb < DIM_; kb += 32) {
    v16bf a = load_a(A + (size_t)rowBase * DIM_ + kb, DIM_, 1);
#pragma unroll
    for (int t = 0; t < 4; ++t) {
      v16bf b = load_b(W + (size_t)kb * DIM_ + colBase + t * 16, DIM_, 1);
      c[t] = wmma_bf16(a, b, c[t]);
    }
  }
#pragma unroll
  for (int t = 0; t < 4; ++t) {
    int col = colBase + t * 16 + c_col();
#pragma unroll
    for (int r = 0; r < 8; ++r) {
      size_t idx = (size_t)(rowBase + c_row(r)) * DIM_ + col;
      float v = alpha * c[t][r];
      if constexpr (STORE_BF16) ((__bf16*)OUT)[idx] = (__bf16)v;
      else                      ((float*)OUT)[idx]  = v + bias[col];
    }
  }
}

// ---------------- 2) diag[row,h] = 0.5 * sum_d s^2 (s already scaled) --------

__global__ void diag_kernel(const __bf16* __restrict__ S, float* __restrict__ diag) {
  int idx = blockIdx.x * blockDim.x + threadIdx.x;  // row*H + h
  if (idx >= ROWS_ * H_) return;
  int row = idx >> 3, h = idx & 7;
  const __bf16* p = S + (size_t)row * DIM_ + h * DH_;
  float s = 0.f;
#pragma unroll 8
  for (int d = 0; d < DH_; ++d) { float v = (float)p[d]; s += v * v; }
  diag[idx] = 0.5f * s;
}

// Shared tile GEMM for feature maps: one 16x16 tile of data_dash for (b,h).
__device__ __forceinline__ v8f dd_tile(const __bf16* Abase, const __bf16* PROJ,
                                       int t) {
  v8f c = {};
#pragma unroll
  for (int kb = 0; kb < DH_; kb += 32) {
    v16bf a  = load_a(Abase + kb, DIM_, 1);
    // element (k,n) = PROJ[(t*16+n)*64 + kb + k]
    v16bf bm = load_b(PROJ + (size_t)(t * 16) * DH_ + kb, 1, DH_);
    c = wmma_bf16(a, bm, c);
  }
  return c;
}

// ---------------- 4) q feature map (per-row max, 2-pass in-register) ---------
// grid (32, 64); block 256; wave -> 16 rows x all 288 cols.

__global__ void phi_q_kernel(const __bf16* __restrict__ QS,
                             const __bf16* __restrict__ PROJ,
                             const float* __restrict__ diag,
                             __bf16* __restrict__ QP, float ratio) {
  int bh = blockIdx.y, b = bh >> 3, h = bh & 7;
  int waveId = threadIdx.x >> 5;
  int nBase  = blockIdx.x * 128 + waveId * 16;
  int lane   = threadIdx.x & 31;
  const __bf16* Abase = QS + ((size_t)b * N_ + nBase) * DIM_ + h * DH_;

  float rmax[8];
#pragma unroll
  for (int r = 0; r < 8; ++r) rmax[r] = -3.4e38f;
  for (int t = 0; t < MP_ / 16; ++t) {           // pass 1: row max (cols < 266)
    v8f c = dd_tile(Abase, PROJ, t);
    if (t * 16 + (lane & 15) < M_) {
#pragma unroll
      for (int r = 0; r < 8; ++r) rmax[r] = fmaxf(rmax[r], c[r]);
    }
  }
#pragma unroll
  for (int m = 1; m < 16; m <<= 1) {             // reduce across 16-lane halves
#pragma unroll
    for (int r = 0; r < 8; ++r) rmax[r] = fmaxf(rmax[r], __shfl_xor(rmax[r], m, 32));
  }
  float dg[8];
#pragma unroll
  for (int r = 0; r < 8; ++r)
    dg[r] = diag[(size_t)(b * N_ + nBase + c_row(r)) * H_ + h];

  for (int t = 0; t < MP_ / 16; ++t) {           // pass 2: recompute + exp
    v8f c = dd_tile(Abase, PROJ, t);
    int col = t * 16 + (lane & 15);
#pragma unroll
    for (int r = 0; r < 8; ++r) {
      int n = nBase + c_row(r);
      float v = (col < M_) ? ratio * (__expf(c[r] - dg[r] - rmax[r]) + 1e-4f) : 0.f;
      QP[((size_t)bh * N_ + n) * MP_ + col] = (__bf16)v;
    }
  }
}

// ---------------- 5a) global max over k data_dash ----------------------------

__global__ void kmax_kernel(const __bf16* __restrict__ KS,
                            const __bf16* __restrict__ PROJ, unsigned* kmax_u) {
  int bh = blockIdx.y, b = bh >> 3, h = bh & 7;
  int waveId = threadIdx.x >> 5;
  int nBase  = blockIdx.x * 128 + waveId * 16;
  int lane   = threadIdx.x & 31;
  const __bf16* Abase = KS + ((size_t)b * N_ + nBase) * DIM_ + h * DH_;
  float mx = -3.4e38f;
  for (int t = 0; t < MP_ / 16; ++t) {
    v8f c = dd_tile(Abase, PROJ, t);
    if (t * 16 + (lane & 15) < M_) {
#pragma unroll
      for (int r = 0; r < 8; ++r) mx = fmaxf(mx, c[r]);
    }
  }
#pragma unroll
  for (int m = 1; m < 32; m <<= 1) mx = fmaxf(mx, __shfl_xor(mx, m, 32));
  __shared__ float red[8];
  if (lane == 0) red[waveId] = mx;
  __syncthreads();
  if (threadIdx.x == 0) {
    float bm = red[0];
#pragma unroll
    for (int i = 1; i < 8; ++i) bm = fmaxf(bm, red[i]);
    atomicMax(kmax_u, f2ord(bm));   // order-preserving -> deterministic
  }
}

// ---------------- 5b) k feature map (single pass, global stab) ---------------

__global__ void phi_k_kernel(const __bf16* __restrict__ KS,
                             const __bf16* __restrict__ PROJ,
                             const float* __restrict__ diag,
                             const unsigned* __restrict__ kmax_u,
                             __bf16* __restrict__ KP, float ratio) {
  int bh = blockIdx.y, b = bh >> 3, h = bh & 7;
  int waveId = threadIdx.x >> 5;
  int nBase  = blockIdx.x * 128 + waveId * 16;
  int lane   = threadIdx.x & 31;
  const __bf16* Abase = KS + ((size_t)b * N_ + nBase) * DIM_ + h * DH_;
  float stab = ord2f(*kmax_u);
  float dg[8];
#pragma unroll
  for (int r = 0; r < 8; ++r)
    dg[r] = diag[(size_t)(b * N_ + nBase + c_row(r)) * H_ + h];
  for (int t = 0; t < MP_ / 16; ++t) {
    v8f c = dd_tile(Abase, PROJ, t);
    int col = t * 16 + (lane & 15);
#pragma unroll
    for (int r = 0; r < 8; ++r) {
      int n = nBase + c_row(r);
      float v = (col < M_) ? ratio * (__expf(c[r] - dg[r] - stab) + 1e-4f) : 0.f;
      KP[((size_t)bh * N_ + n) * MP_ + col] = (__bf16)v;
    }
  }
}

// ---------------- 6a) ksum[b,h,m] = sum_n kp --------------------------------

__global__ void ksum_kernel(const __bf16* __restrict__ KP, float* __restrict__ ksum) {
  int bh = blockIdx.x;
  int m  = threadIdx.x;          // blockDim == MP_
  if (m >= MP_) return;
  const __bf16* p = KP + (size_t)bh * N_ * MP_ + m;
  float s = 0.f;
  for (int n = 0; n < N_; ++n) s += (float)p[(size_t)n * MP_];
  ksum[bh * MP_ + m] = s;
}

// ---------------- 6b) context[b,h,m,d] = kp^T @ v (K = 4096) -----------------
// grid (9, 64); block 256: 72 wave-tiles = 18 mtiles x 4 ntiles per (b,h).

__global__ void context_kernel(const __bf16* __restrict__ KP,
                               const __bf16* __restrict__ VB,
                               __bf16* __restrict__ CTX) {
  int bh = blockIdx.y, b = bh >> 3, h = bh & 7;
  int tile = blockIdx.x * 8 + (threadIdx.x >> 5);
  int mt = tile >> 2, nt = tile & 3;
  const __bf16* Ab = KP + (size_t)bh * N_ * MP_ + mt * 16;          // (m,k)->Ab[k*MP_+m]
  const __bf16* Bb = VB + (size_t)b * N_ * DIM_ + h * DH_ + nt * 16; // (k,n)->Bb[k*DIM_+n]
  v8f c = {};
  for (int kb = 0; kb < N_; kb += 32) {
    v16bf a  = load_a(Ab + (size_t)kb * MP_, 1, MP_);
    v16bf bm = load_b(Bb + (size_t)kb * DIM_, DIM_, 1);
    c = wmma_bf16(a, bm, c);
  }
  int col = nt * 16 + c_col();
#pragma unroll
  for (int r = 0; r < 8; ++r) {
    int m = mt * 16 + c_row(r);
    CTX[(size_t)bh * MP_ * DH_ + (size_t)m * DH_ + col] = (__bf16)c[r];
  }
}

// ---------------- 7a) dinv = 1/(qp . ksum), ksum staged in LDS ---------------

__global__ void dinv_kernel(const __bf16* __restrict__ QP,
                            const float* __restrict__ ksum,
                            float* __restrict__ dinv) {
  int bh = blockIdx.x;
  __shared__ float sk[MP_];
  for (int i = threadIdx.x; i < MP_; i += blockDim.x) sk[i] = ksum[bh * MP_ + i];
  __syncthreads();
  for (int n = threadIdx.x; n < N_; n += blockDim.x) {
    const __bf16* p = QP + ((size_t)bh * N_ + n) * MP_;
    float s = 0.f;
#pragma unroll 8
    for (int m = 0; m < MP_; ++m) s += (float)p[m] * sk[m];
    dinv[(size_t)bh * N_ + n] = 1.0f / s;
  }
}

// ---------------- 7b) attn = (qp @ context) * dinv  -> bf16 [row,512] --------
// grid (32, 64); block 256; wave -> 16 rows x 64 cols; K = 288.

__global__ void attn_out_kernel(const __bf16* __restrict__ QP,
                                const __bf16* __restrict__ CTX,
                                const float* __restrict__ dinv,
                                __bf16* __restrict__ ATTN) {
  int bh = blockIdx.y, b = bh >> 3, h = bh & 7;
  int waveId = threadIdx.x >> 5;
  int nBase  = blockIdx.x * 128 + waveId * 16;
  const __bf16* Ab = QP + ((size_t)bh * N_ + nBase) * MP_;
  const __bf16* Cb = CTX + (size_t)bh * MP_ * DH_;
  v8f c[4] = {};
  for (int kb = 0; kb < MP_; kb += 32) {
    v16bf a = load_a(Ab + kb, MP_, 1);
#pragma unroll
    for (int t = 0; t < 4; ++t) {
      v16bf bm = load_b(Cb + (size_t)kb * DH_ + t * 16, DH_, 1);
      c[t] = wmma_bf16(a, bm, c[t]);
    }
  }
  float di[8];
#pragma unroll
  for (int r = 0; r < 8; ++r) di[r] = dinv[(size_t)bh * N_ + nBase + c_row(r)];
#pragma unroll
  for (int t = 0; t < 4; ++t) {
    int d = t * 16 + c_col();
#pragma unroll
    for (int r = 0; r < 8; ++r) {
      int n = nBase + c_row(r);
      ATTN[((size_t)b * N_ + n) * DIM_ + h * DH_ + d] = (__bf16)(c[t][r] * di[r]);
    }
  }
}

// ---------------- host launch ----------------

extern "C" void kernel_launch(void* const* d_in, const int* in_sizes, int n_in,
                              void* d_out, int out_size, void* d_ws, size_t ws_size,
                              hipStream_t stream) {
  const float* x    = (const float*)d_in[0];
  const float* Wq   = (const float*)d_in[1];
  const float* Wk   = (const float*)d_in[2];
  const float* Wv   = (const float*)d_in[3];
  const float* Wo   = (const float*)d_in[4];
  const float* bo   = (const float*)d_in[5];
  const float* proj = (const float*)d_in[6];
  float* out = (float*)d_out;

  char* ws = (char*)d_ws;
  size_t off = 0;
  auto take = [&](size_t bytes) {
    size_t o = off;
    off += (bytes + 255) & ~(size_t)255;
    return o;
  };
  // ~452 MB total workspace.
  __bf16*   xb     = (__bf16*)(ws + take((size_t)ROWS_ * DIM_ * 2));
  __bf16*   wq_b   = (__bf16*)(ws + take((size_t)DIM_ * DIM_ * 2));
  __bf16*   wk_b   = (__bf16*)(ws + take((size_t)DIM_ * DIM_ * 2));
  __bf16*   wv_b   = (__bf16*)(ws + take((size_t)DIM_ * DIM_ * 2));
  __bf16*   wo_b   = (__bf16*)(ws + take((size_t)DIM_ * DIM_ * 2));
  __bf16*   projb  = (__bf16*)(ws + take((size_t)MP_ * DH_ * 2));
  __bf16*   qs_b   = (__bf16*)(ws + take((size_t)ROWS_ * DIM_ * 2));
  __bf16*   ks_b   = (__bf16*)(ws + take((size_t)ROWS_ * DIM_ * 2));
  __bf16*   vb     = (__bf16*)(ws + take((size_t)ROWS_ * DIM_ * 2));
  float*    diag_q = (float*)(ws + take((size_t)ROWS_ * H_ * 4));
  float*    diag_k = (float*)(ws + take((size_t)ROWS_ * H_ * 4));
  unsigned* kmax_u = (unsigned*)(ws + take(256));
  __bf16*   qp_b   = (__bf16*)(ws + take((size_t)B_ * H_ * N_ * MP_ * 2));
  __bf16*   kp_b   = (__bf16*)(ws + take((size_t)B_ * H_ * N_ * MP_ * 2));
  float*    ksum   = (float*)(ws + take((size_t)B_ * H_ * MP_ * 4));
  __bf16*   ctx_b  = (__bf16*)(ws + take((size_t)B_ * H_ * MP_ * DH_ * 2));
  float*    dinv   = (float*)(ws + take((size_t)B_ * H_ * N_ * 4));
  __bf16*   attn_b = xb;  // x_bf16 is dead after the projection GEMMs

  const float normalizer = 0.35355339059327373f;  // 64^-0.25
  const float ratio      = 0.06131393076332176f;  // 266^-0.5

  // 1) casts
  {
    size_t n = (size_t)ROWS_ * DIM_;
    cast_bf16_kernel<<<(unsigned)((n + 255) / 256), 256, 0, stream>>>(x, xb, n);
    size_t wN = (size_t)DIM_ * DIM_;
    unsigned wg = (unsigned)((wN + 255) / 256);
    cast_bf16_kernel<<<wg, 256, 0, stream>>>(Wq, wq_b, wN);
    cast_bf16_kernel<<<wg, 256, 0, stream>>>(Wk, wk_b, wN);
    cast_bf16_kernel<<<wg, 256, 0, stream>>>(Wv, wv_b, wN);
    cast_bf16_kernel<<<wg, 256, 0, stream>>>(Wo, wo_b, wN);
    proj_pack_kernel<<<(MP_ * DH_ + 255) / 256, 256, 0, stream>>>(proj, projb);
    init_kmax_kernel<<<1, 32, 0, stream>>>(kmax_u);
  }

  // 2) projections (normalizer folded into q,k)
  dim3 gP(ROWS_ / 128, DIM_ / 64);
  proj_gemm_kernel<true><<<gP, 256, 0, stream>>>(xb, wq_b, qs_b, nullptr, normalizer);
  proj_gemm_kernel<true><<<gP, 256, 0, stream>>>(xb, wk_b, ks_b, nullptr, normalizer);
  proj_gemm_kernel<true><<<gP, 256, 0, stream>>>(xb, wv_b, vb, nullptr, 1.0f);

  // 3) diagonals
  diag_kernel<<<(ROWS_ * H_) / 256, 256, 0, stream>>>(qs_b, diag_q);
  diag_kernel<<<(ROWS_ * H_) / 256, 256, 0, stream>>>(ks_b, diag_k);

  // 4) feature maps
  dim3 gF(N_ / 128, B_ * H_);
  phi_q_kernel<<<gF, 256, 0, stream>>>(qs_b, projb, diag_q, qp_b, ratio);
  kmax_kernel<<<gF, 256, 0, stream>>>(ks_b, projb, kmax_u);
  phi_k_kernel<<<gF, 256, 0, stream>>>(ks_b, projb, diag_k, kmax_u, kp_b, ratio);

  // 5) linear attention
  ksum_kernel<<<B_ * H_, MP_, 0, stream>>>(kp_b, ksum);
  context_kernel<<<dim3(9, B_ * H_), 256, 0, stream>>>(kp_b, vb, ctx_b);
  dinv_kernel<<<B_ * H_, 256, 0, stream>>>(qp_b, ksum, dinv);
  attn_out_kernel<<<gF, 256, 0, stream>>>(qp_b, ctx_b, dinv, attn_b);

  // 6) output projection + bias (f32 out)
  proj_gemm_kernel<false><<<gP, 256, 0, stream>>>(attn_b, wo_b, (void*)out, bo, 1.0f);
}
